// WassersteinLoss_90769838834213
// MI455X (gfx1250) — matrix operand order repaired and checked by reading, not compiled, not run on hardware
//
#include <hip/hip_runtime.h>
#include <hip/hip_bf16.h>

#define BB 512
#define CC 1024

typedef __attribute__((ext_vector_type(16))) __bf16 v16bf;
typedef __attribute__((ext_vector_type(8)))  __bf16 v8bf;
typedef __attribute__((ext_vector_type(8)))  float  v8f;

// ---------------------------------------------------------------------------
// Kernel 1: per-row softmax (fp32), mask (bf16), 1/count (fp32).
// One block (256 threads) per row; deterministic tree reductions in LDS.
// ---------------------------------------------------------------------------
__global__ void wl_prep(const float* __restrict__ logits,
                        const int*   __restrict__ targets,
                        float* __restrict__ P,
                        __bf16* __restrict__ mask,
                        float* __restrict__ invc) {
    __shared__ float red[256];
    const int r = blockIdx.x;
    const int t = threadIdx.x;

    float x[4];
    float mx = -3.402823466e+38f;
#pragma unroll
    for (int i = 0; i < 4; ++i) {
        x[i] = logits[r * CC + t + i * 256];
        mx = fmaxf(mx, x[i]);
    }
    red[t] = mx; __syncthreads();
    for (int s = 128; s > 0; s >>= 1) {
        if (t < s) red[t] = fmaxf(red[t], red[t + s]);
        __syncthreads();
    }
    mx = red[0]; __syncthreads();

    float e[4], sum = 0.f;
#pragma unroll
    for (int i = 0; i < 4; ++i) { e[i] = __expf(x[i] - mx); sum += e[i]; }
    red[t] = sum; __syncthreads();
    for (int s = 128; s > 0; s >>= 1) {
        if (t < s) red[t] += red[t + s];
        __syncthreads();
    }
    const float inv = 1.0f / red[0]; __syncthreads();
#pragma unroll
    for (int i = 0; i < 4; ++i) P[r * CC + t + i * 256] = e[i] * inv;

    float cnt = 0.f;
#pragma unroll
    for (int i = 0; i < 4; ++i) {
        const int idx = t + i * 256;
        const float m = (targets[r * CC + idx] != 0) ? 1.0f : 0.0f;
        mask[r * CC + idx] = (__bf16)m;   // 0/1 exact in bf16
        cnt += m;
    }
    red[t] = cnt; __syncthreads();
    for (int s = 128; s > 0; s >>= 1) {
        if (t < s) red[t] += red[t + s];
        __syncthreads();
    }
    if (t == 0) invc[r] = 1.0f / red[0];
}

// ---------------------------------------------------------------------------
// Kernel 2: split D (fp32) into bf16 hi/lo, pre-packed into this file's WMMA
// B-operand order (contiguous k = 16h + s, not AMD's table order; 16x16x32 bf16):
//   lane = n%16 + 16*(k%32 >= 16); element-in-lane = k%16
//   (element 2j -> low half of VGPR j, 2j+1 -> high half — matches memory order)
// ---------------------------------------------------------------------------
__global__ void wl_pack_d(const float* __restrict__ D,
                          __bf16* __restrict__ dhi,
                          __bf16* __restrict__ dlo) {
    const int idx = blockIdx.x * 256 + threadIdx.x;  // over CC*CC
    const int k = idx >> 10;
    const int n = idx & (CC - 1);
    const float d = D[idx];
    const __bf16 hi = (__bf16)d;
    const __bf16 lo = (__bf16)(d - (float)hi);
    const int ktile = k >> 5, kr = k & 31;
    const int ntile = n >> 4, nc = n & 15;
    const int lane  = nc | (kr & 16);
    const int kl    = kr & 15;
    const size_t off = ((size_t)(ktile * (CC / 16) + ntile) * 32 + lane) * 16 + kl;
    dhi[off] = hi;
    dlo[off] = lo;
}

// ---------------------------------------------------------------------------
// Kernel 3: fused GEMM + loss. Block = 128 threads (4 waves), 32x32 tile,
// each wave owns one 16x16 output tile; K loop of 32 with 2 WMMAs (hi+lo).
// Then weight by P[m,n]*invc[m] in registers and reduce to one block partial.
// ---------------------------------------------------------------------------
__global__ void wl_gemm_loss(const __bf16* __restrict__ mask,
                             const __bf16* __restrict__ dhi,
                             const __bf16* __restrict__ dlo,
                             const float*  __restrict__ P,
                             const float*  __restrict__ invc,
                             float* __restrict__ partials) {
    __shared__ float wsum[4];
    const int w    = threadIdx.x >> 5;   // wave 0..3
    const int lane = threadIdx.x & 31;
    const int m0 = blockIdx.y * 32 + (w >> 1) * 16;
    const int n0 = blockIdx.x * 32 + (w & 1) * 16;
    const int ntile = n0 >> 4;
    const int sel = lane >> 4;           // 0: lanes 0-15, 1: lanes 16-31

    // A layout (16x32 bf16): lane holds row M=lane&15;
// elements 0..15 = K(kk + 16*sel .. +15)
    const __bf16* arow = mask + (size_t)(m0 + (lane & 15)) * CC;

    v8f c = {};
    for (int kk = 0; kk < CC; kk += 32) {
        union { v16bf v; v8bf h[2]; } a;
        a.h[0] = *(const v8bf*)(arow + kk + sel * 16);
        a.h[1] = *(const v8bf*)(arow + kk + sel * 16 + 8);

        const int kt = kk >> 5;
        const size_t boff = ((size_t)(kt * (CC / 16) + ntile) * 32 + lane) * 16;
        const v16bf bh = *(const v16bf*)(dhi + boff);
        const v16bf bl = *(const v16bf*)(dlo + boff);

        c = __builtin_amdgcn_wmma_f32_16x16x32_bf16(false, a.v, false, bh,
                                                    (short)0, c, false, false);
        c = __builtin_amdgcn_wmma_f32_16x16x32_bf16(false, a.v, false, bl,
                                                    (short)0, c, false, false);
    }

    // C layout: VGPR r, this lane -> M = m0 + r + 8*sel, N = n0 + (lane&15)
    const int ncol = n0 + (lane & 15);
    float acc = 0.f;
#pragma unroll
    for (int r = 0; r < 8; ++r) {
        const int m = m0 + r + 8 * sel;
        acc += c[r] * P[(size_t)m * CC + ncol] * invc[m];
    }
    // wave32 reduce (fixed order -> deterministic)
    for (int off = 16; off > 0; off >>= 1)
        acc += __shfl_xor(acc, off, 32);
    if (lane == 0) wsum[w] = acc;
    __syncthreads();
    if (threadIdx.x == 0)
        partials[blockIdx.y * gridDim.x + blockIdx.x] =
            wsum[0] + wsum[1] + wsum[2] + wsum[3];
}

// ---------------------------------------------------------------------------
// Kernel 4: deterministic final reduction of 512 partials; mean over B.
// ---------------------------------------------------------------------------
__global__ void wl_finalize(const float* __restrict__ partials,
                            float* __restrict__ out) {
    __shared__ float red[256];
    const int t = threadIdx.x;
    red[t] = partials[t] + partials[t + 256];
    __syncthreads();
    for (int s = 128; s > 0; s >>= 1) {
        if (t < s) red[t] += red[t + s];
        __syncthreads();
    }
    if (t == 0) out[0] = red[0] / (float)BB;
}

extern "C" void kernel_launch(void* const* d_in, const int* in_sizes, int n_in,
                              void* d_out, int out_size, void* d_ws, size_t ws_size,
                              hipStream_t stream) {
    const float* logits  = (const float*)d_in[0];
    const int*   targets = (const int*)d_in[1];
    const float* D       = (const float*)d_in[2];

    char* ws = (char*)d_ws;
    float*  P        = (float*)(ws);                       // 512*1024*4 = 2 MB
    __bf16* maskbf   = (__bf16*)(ws + 2097152);            // 1 MB
    float*  invc     = (float*)(ws + 3145728);             // 2 KB
    __bf16* dhi      = (__bf16*)(ws + 3147776);            // 2 MB (packed)
    __bf16* dlo      = (__bf16*)(ws + 5244928);            // 2 MB (packed)
    float*  partials = (float*)(ws + 7342080);             // 512*4 = 2 KB

    wl_prep<<<BB, 256, 0, stream>>>(logits, targets, P, maskbf, invc);
    wl_pack_d<<<(CC * CC) / 256, 256, 0, stream>>>(D, dhi, dlo);

    dim3 grid(CC / 32, BB / 32);   // (32, 16) = 512 blocks of 4 waves
    wl_gemm_loss<<<grid, 128, 0, stream>>>(maskbf, dhi, dlo, P, invc, partials);

    wl_finalize<<<1, 256, 0, stream>>>(partials, (float*)d_out);
}
